// IncrementalResNet18V2_9689446220303
// MI455X (gfx1250) — compile-verified, matrix operand order checked by reading, not run on hardware
//
#include <hip/hip_runtime.h>
#include <hip/hip_bf16.h>

// ---------------------------------------------------------------------------
// ResNet18 inference for gfx1250 (MI455X): bf16 NHWC activations, WMMA
// implicit-GEMM convolutions (v_wmma_f32_16x16x32_bf16), fused BN/ReLU/residual.
// Fast path: 4-wave workgroups, B fragments staged in double-buffered LDS
// (ds_load/ds_store + workgroup barrier), div-free flat k-tile loop.
// ---------------------------------------------------------------------------

typedef __attribute__((ext_vector_type(16))) __bf16 v16bf;
typedef __attribute__((ext_vector_type(8)))  __bf16 v8bf;
typedef __attribute__((ext_vector_type(8)))  float  v8f;

#define BN_EPS 1e-5f

// K-position of element e of a v16bf A/B fragment, per CDNA5 ISA 16-bit layout:
// elements 0..7  -> K = e      + 8*hi
// elements 8..15 -> K = e + 8  + 8*hi   (i.e. K=16..23 (+8*hi))
__device__ __forceinline__ int klocal(int e, int hi) {
  int v = e >> 1;
  int k = (v < 4) ? (2 * v + (e & 1)) : (16 + 2 * (v - 4) + (e & 1));
  return k + (hi ? 8 : 0);
}

union AFrag { v16bf v; v8bf run[2]; __bf16 h[16]; };
union CFrag { v8f   v; float f[8]; };

// ---------------------------------------------------------------------------
// Weight repack: [Cout,Cin,KH,KW] fp32 -> bf16 B-fragments
// layout [Kpad/32][ntiles][lane 0..31][e 0..15]; lane&15 = N column,
// K-half selected by lane>=16, zero-padded past K. K order = (kh,kw,cin).
// ---------------------------------------------------------------------------
template<int CIN, int KH, int KW>
__global__ void pack_w(const float* __restrict__ w, __bf16* __restrict__ dst,
                       int Cout, int ntiles) {
  constexpr int K    = CIN * KH * KW;
  constexpr int Kpad = (K + 31) & ~31;
  const size_t total = (size_t)(Kpad >> 5) * ntiles * 512;
  size_t idx = (size_t)blockIdx.x * blockDim.x + threadIdx.x;
  if (idx >= total) return;
  int e     = (int)(idx & 15);
  int lane  = (int)((idx >> 4) & 31);
  size_t t2 = idx >> 9;
  int ntile = (int)(t2 % ntiles);
  int ktile = (int)(t2 / ntiles);
  int n = ntile * 16 + (lane & 15);
  int k = ktile * 32 + klocal(e, lane >= 16);
  float v = 0.f;
  if (n < Cout && k < K) {
    int cin = k % CIN; int t = k / CIN;
    int kw = t % KW;   int kh = t / KW;
    v = w[(((size_t)n * CIN + cin) * KH + kh) * KW + kw];
  }
  dst[idx] = (__bf16)v;
}

// ---------------------------------------------------------------------------
// FAST implicit-GEMM conv + BN (+residual)(+ReLU), CIN % 32 == 0.
// 4-wave workgroup; each wave owns a 32(M) x 64(N) output tile (2 M x 4 N
// 16x16 WMMA tiles); all waves share the same 64 columns. Per k-tile the
// block cooperatively stages the 4KB B chunk into double-buffered LDS
// (128 threads x 32B), then each wave broadcast-reads its fragments.
// ---------------------------------------------------------------------------
template<int CIN, int KHW>
__global__ __launch_bounds__(128)
void conv_bn_wmma_fast(const __bf16* __restrict__ in, const __bf16* __restrict__ wpk,
                       const float* __restrict__ bn_mean, const float* __restrict__ bn_var,
                       const float* __restrict__ bn_gamma, const float* __restrict__ bn_beta,
                       const __bf16* __restrict__ res, __bf16* __restrict__ out,
                       int Nb, int H, int W, int OH, int OW, int Cout,
                       int stride, int pad, int relu, int ntiles) {
  constexpr int CBLK = CIN / 32;                 // k-tiles per tap (pow2: 2..16)
  constexpr int CSH  = (CBLK == 2) ? 1 : (CBLK == 4) ? 2 : (CBLK == 8) ? 3 : 4;
  constexpr int NKT  = KHW * KHW * CBLK;         // total k-tiles

  __shared__ __align__(32) __bf16 smem[2][4 * 512];   // 2 x 4KB B double buffer

  const int tid  = threadIdx.x;                  // 0..127
  const int lane = tid & 31;
  const int wv   = tid >> 5;                     // wave 0..3
  const int hi   = lane >> 4;
  const int mtw  = blockIdx.x * 4 + wv;          // per-wave 32-row supertile
  const int ng   = blockIdx.y;                   // group of 4 N-tiles
  const int M    = Nb * OH * OW;

  int nn[2], ohv[2], owv[2]; bool rv[2];
  #pragma unroll
  for (int i = 0; i < 2; i++) {
    int row = mtw * 32 + i * 16 + (lane & 15);
    rv[i] = row < M;
    int r = rv[i] ? row : 0;
    nn[i] = r / (OH * OW); int rr = r % (OH * OW);
    ohv[i] = rr / OW; owv[i] = rr % OW;
  }

  CFrag acc[2][4];
  #pragma unroll
  for (int i = 0; i < 2; i++)
    #pragma unroll
    for (int j = 0; j < 4; j++)
      #pragma unroll
      for (int q = 0; q < 8; q++) acc[i][j].f[q] = 0.f;

  AFrag zf;
  #pragma unroll
  for (int e = 0; e < 16; e++) zf.h[e] = (__bf16)0.f;

  // cooperative B stage: 128 threads x 32B = 4KB chunk (4 N-tiles) per k-tile
  auto stageB = [&](int kt, int buf) {
    const __bf16* g = wpk + ((size_t)kt * ntiles + (size_t)ng * 4) * 512 + tid * 16;
    *(v16bf*)&smem[buf][tid * 16] = *(const v16bf*)g;
  };
  auto loadA = [&](int kt, AFrag (&a)[2]) {
    int tap = kt >> CSH;                         // shifts/masks only
    int c0  = (kt & (CBLK - 1)) << 5;
    int kh  = tap / KHW;                         // constant divisor
    int kw  = tap % KHW;
    #pragma unroll
    for (int i = 0; i < 2; i++) {
      int ih = ohv[i] * stride - pad + kh;
      int iw = owv[i] * stride - pad + kw;
      bool ok = rv[i] && ih >= 0 && ih < H && iw >= 0 && iw < W;
      if (ok) {
        const __bf16* p = in + (((size_t)nn[i] * H + ih) * W + iw) * CIN + c0 + 8 * hi;
        a[i].run[0] = *(const v8bf*)p;
        a[i].run[1] = *(const v8bf*)(p + 16);
      } else {
        a[i] = zf;
      }
    }
  };

  int buf = 0;
  stageB(0, 0);
  __syncthreads();

  #pragma unroll 1
  for (int kt = 0; kt < NKT; kt++) {
    if (kt + 1 < NKT) stageB(kt + 1, buf ^ 1);   // prefetch next chunk into other buffer

    AFrag a[2];
    loadA(kt, a);
    v16bf b[4];
    #pragma unroll
    for (int j = 0; j < 4; j++)
      b[j] = *(const v16bf*)&smem[buf][j * 512 + lane * 16];   // ds_load broadcast

    #pragma unroll
    for (int i = 0; i < 2; i++)
      #pragma unroll
      for (int j = 0; j < 4; j++)
        acc[i][j].v = __builtin_amdgcn_wmma_f32_16x16x32_bf16(
            false, a[i].v, false, b[j], (short)0, acc[i][j].v, false, false);

    __syncthreads();                             // buf reads done, buf^1 staged
    buf ^= 1;
  }

  // --- epilogue: BN fold, residual, ReLU, bf16 NHWC store ---
  #pragma unroll
  for (int j = 0; j < 4; j++) {
    int ncol = (ng * 4 + j) * 16 + (lane & 15);
    float sc = bn_gamma[ncol] * rsqrtf(bn_var[ncol] + BN_EPS);
    float bi = bn_beta[ncol] - bn_mean[ncol] * sc;
    #pragma unroll
    for (int i = 0; i < 2; i++) {
      #pragma unroll
      for (int r8 = 0; r8 < 8; r8++) {
        int m = mtw * 32 + i * 16 + r8 + (hi ? 8 : 0);
        if (m < M) {
          float v = acc[i][j].f[r8] * sc + bi;
          if (res)  v += (float)res[(size_t)m * Cout + ncol];
          if (relu) v = fmaxf(v, 0.f);
          out[(size_t)m * Cout + ncol] = (__bf16)v;
        }
      }
    }
  }
}

// ---------------------------------------------------------------------------
// Generic conv (conv1 only: Cin=3, 7x7): per-element im2col gather with
// compile-time divisors. One wave per 16(M) x 64(N) tile.
// ---------------------------------------------------------------------------
template<int CIN, int KH, int KW>
__global__ __launch_bounds__(32)
void conv_bn_wmma_gen(const __bf16* __restrict__ in, const __bf16* __restrict__ wpk,
                      const float* __restrict__ bn_mean, const float* __restrict__ bn_var,
                      const float* __restrict__ bn_gamma, const float* __restrict__ bn_beta,
                      __bf16* __restrict__ out,
                      int Nb, int H, int W, int OH, int OW, int Cout,
                      int stride, int pad, int relu, int ntiles) {
  constexpr int K    = CIN * KH * KW;
  constexpr int Kpad = (K + 31) & ~31;

  const int lane = threadIdx.x & 31;
  const int hi   = lane >> 4;
  const int mt   = blockIdx.x;
  const int ng   = blockIdx.y;
  const int M    = Nb * OH * OW;
  const int row  = mt * 16 + (lane & 15);
  const bool rowValid = row < M;
  int n = 0, oh = 0, ow = 0;
  if (rowValid) { n = row / (OH * OW); int r = row % (OH * OW); oh = r / OW; ow = r % OW; }

  CFrag acc[4];
  #pragma unroll
  for (int j = 0; j < 4; j++)
    #pragma unroll
    for (int q = 0; q < 8; q++) acc[j].f[q] = 0.f;

  #pragma unroll 1
  for (int k0 = 0; k0 < Kpad; k0 += 32) {
    AFrag a;
    #pragma unroll
    for (int e = 0; e < 16; e++) {
      int k = k0 + klocal(e, hi);
      float vf = 0.f;
      if (rowValid && k < K) {
        int cin = k % CIN; int t = k / CIN;     // constant divisors
        int kw = t % KW;   int kh = t / KW;
        int ih = oh * stride - pad + kh;
        int iw = ow * stride - pad + kw;
        if (ih >= 0 && ih < H && iw >= 0 && iw < W)
          vf = (float)in[(((size_t)n * H + ih) * W + iw) * CIN + cin];
      }
      a.h[e] = (__bf16)vf;
    }
    const __bf16* bb = wpk + ((size_t)(k0 >> 5) * ntiles + (size_t)ng * 4) * 512 + lane * 16;
    v16bf bfr[4];
    #pragma unroll
    for (int j = 0; j < 4; j++) bfr[j] = *(const v16bf*)(bb + j * 512);
    #pragma unroll
    for (int j = 0; j < 4; j++)
      acc[j].v = __builtin_amdgcn_wmma_f32_16x16x32_bf16(
          false, a.v, false, bfr[j], (short)0, acc[j].v, false, false);
  }

  #pragma unroll
  for (int j = 0; j < 4; j++) {
    int ncol = (ng * 4 + j) * 16 + (lane & 15);
    float sc = bn_gamma[ncol] * rsqrtf(bn_var[ncol] + BN_EPS);
    float bi = bn_beta[ncol] - bn_mean[ncol] * sc;
    #pragma unroll
    for (int r8 = 0; r8 < 8; r8++) {
      int m = mt * 16 + r8 + (hi ? 8 : 0);
      if (m < M) {
        float v = acc[j].f[r8] * sc + bi;
        if (relu) v = fmaxf(v, 0.f);
        out[(size_t)m * Cout + ncol] = (__bf16)v;
      }
    }
  }
}

// ---------------------------------------------------------------------------
// FC: [32,512] x [512,1000] + bias -> fp32 d_out. N padded to 1008.
// ---------------------------------------------------------------------------
__global__ __launch_bounds__(32)
void fc_wmma(const __bf16* __restrict__ a, const __bf16* __restrict__ wpk,
             const float* __restrict__ bias, float* __restrict__ out,
             int M, int Kdim, int Nout, int ntiles) {
  const int lane = threadIdx.x & 31;
  const int hi   = lane >> 4;
  const int mt   = blockIdx.x;
  const int nt   = blockIdx.y;
  const int row  = mt * 16 + (lane & 15);
  const bool rowValid = row < M;
  const __bf16* abase = a + (size_t)(rowValid ? row : 0) * Kdim + 8 * hi;

  CFrag acc;
  #pragma unroll
  for (int q = 0; q < 8; q++) acc.f[q] = 0.f;
  AFrag zf;
  #pragma unroll
  for (int e = 0; e < 16; e++) zf.h[e] = (__bf16)0.f;

  #pragma unroll 1
  for (int k0 = 0; k0 < Kdim; k0 += 32) {
    AFrag af;
    if (rowValid) {
      af.run[0] = *(const v8bf*)(abase + k0);
      af.run[1] = *(const v8bf*)(abase + k0 + 16);
    } else {
      af = zf;
    }
    v16bf bf = *(const v16bf*)(wpk + ((size_t)(k0 >> 5) * ntiles + nt) * 512 + lane * 16);
    acc.v = __builtin_amdgcn_wmma_f32_16x16x32_bf16(
        false, af.v, false, bf, (short)0, acc.v, false, false);
  }

  int ncol = nt * 16 + (lane & 15);
  if (ncol < Nout) {
    float b = bias[ncol];
    #pragma unroll
    for (int r8 = 0; r8 < 8; r8++) {
      int m = mt * 16 + r8 + (hi ? 8 : 0);
      if (m < M) out[(size_t)m * Nout + ncol] = acc.f[r8] + b;
    }
  }
}

// ---------------------------------------------------------------------------
// Layout / pooling helpers.
// ---------------------------------------------------------------------------
__global__ void nchw_to_nhwc(const float* __restrict__ x, __bf16* __restrict__ out,
                             int Nb, int C, int H, int W) {
  size_t idx = (size_t)blockIdx.x * blockDim.x + threadIdx.x;
  size_t total = (size_t)Nb * C * H * W;
  if (idx >= total) return;
  int c = (int)(idx % C); size_t t = idx / C;
  int w = (int)(t % W); t /= W;
  int h = (int)(t % H); int n = (int)(t / H);
  out[idx] = (__bf16)x[(((size_t)n * C + c) * H + h) * W + w];
}

__global__ void maxpool3s2_k(const __bf16* __restrict__ in, __bf16* __restrict__ out,
                             int Nb, int H, int W, int C, int OH, int OW) {
  size_t idx = (size_t)blockIdx.x * blockDim.x + threadIdx.x;
  size_t total = (size_t)Nb * OH * OW * C;
  if (idx >= total) return;
  int c = (int)(idx % C); size_t t = idx / C;
  int ow = (int)(t % OW); t /= OW;
  int oh = (int)(t % OH); int n = (int)(t / OH);
  float m = -3.4e38f;
  for (int dh = 0; dh < 3; dh++)
    for (int dw = 0; dw < 3; dw++) {
      int ih = oh * 2 - 1 + dh, iw = ow * 2 - 1 + dw;
      if (ih >= 0 && ih < H && iw >= 0 && iw < W)
        m = fmaxf(m, (float)in[(((size_t)n * H + ih) * W + iw) * C + c]);
    }
  out[idx] = (__bf16)m;
}

__global__ void avgpool7(const __bf16* __restrict__ in, __bf16* __restrict__ out,
                         int Nb, int C) {
  size_t idx = (size_t)blockIdx.x * blockDim.x + threadIdx.x;
  if (idx >= (size_t)Nb * C) return;
  int c = (int)(idx % C); int n = (int)(idx / C);
  float s = 0.f;
  for (int i = 0; i < 49; i++) s += (float)in[((size_t)n * 49 + i) * C + c];
  out[idx] = (__bf16)(s * (1.f / 49.f));
}

// ---------------------------------------------------------------------------
// Host-side templated layer launcher (fast path, CIN % 32 == 0).
// Param group base index -> {w, mean, var, gamma, beta}.
// ---------------------------------------------------------------------------
template<int CIN, int KHW>
static inline void conv_layer(void* const* d_in, int base, const __bf16* in,
                              int H, int W, int Cout, int stride, int pad,
                              const __bf16* res, int relu,
                              __bf16* outp, __bf16* packb, hipStream_t stream) {
  const int K = CIN * KHW * KHW;
  const int ntiles = Cout / 16;
  const size_t packElems = (size_t)(K >> 5) * ntiles * 512;
  pack_w<CIN, KHW, KHW><<<dim3((unsigned)((packElems + 255) / 256)), 256, 0, stream>>>(
      (const float*)d_in[base], packb, Cout, ntiles);
  const int OH = (H + 2 * pad - KHW) / stride + 1;
  const int OW = (W + 2 * pad - KHW) / stride + 1;
  const int M = 32 * OH * OW;
  dim3 grid((unsigned)((M + 127) / 128), (unsigned)(Cout / 64));
  conv_bn_wmma_fast<CIN, KHW><<<grid, 128, 0, stream>>>(
      in, packb,
      (const float*)d_in[base + 1], (const float*)d_in[base + 2],
      (const float*)d_in[base + 3], (const float*)d_in[base + 4],
      res, outp, 32, H, W, OH, OW, Cout, stride, pad, relu, ntiles);
}

// ---------------------------------------------------------------------------
// Host driver
// ---------------------------------------------------------------------------
extern "C" void kernel_launch(void* const* d_in, const int* in_sizes, int n_in,
                              void* d_out, int out_size, void* d_ws, size_t ws_size,
                              hipStream_t stream) {
  (void)in_sizes; (void)n_in; (void)out_size; (void)ws_size;

  // ---- workspace layout (bytes) ----
  constexpr size_t OFF_BX   = 0;                                   // 32*224*224*3 bf16
  constexpr size_t OFF_ACT0 = OFF_BX   + 32ull*224*224*3*2;        // 25.69M bf16 (big)
  constexpr size_t OFF_ACT1 = OFF_ACT0 + 32ull*112*112*64*2;       // 6.42M bf16
  constexpr size_t OFF_ACT2 = OFF_ACT1 + 32ull*56*56*64*2;
  constexpr size_t OFF_ACT3 = OFF_ACT2 + 32ull*56*56*64*2;
  constexpr size_t OFF_POOL = OFF_ACT3 + 32ull*56*56*64*2;         // 32*512 bf16
  constexpr size_t OFF_PACK = OFF_POOL + 32ull*512*2;              // 4608*512 bf16 max

  char* ws = (char*)d_ws;
  __bf16* bx     = (__bf16*)(ws + OFF_BX);
  __bf16* act0   = (__bf16*)(ws + OFF_ACT0);
  __bf16* act1   = (__bf16*)(ws + OFF_ACT1);
  __bf16* act2   = (__bf16*)(ws + OFF_ACT2);
  __bf16* act3   = (__bf16*)(ws + OFF_ACT3);
  __bf16* pooled = (__bf16*)(ws + OFF_POOL);
  __bf16* packb  = (__bf16*)(ws + OFF_PACK);

  // ---- input: fp32 NCHW -> bf16 NHWC ----
  {
    size_t total = 32ull * 3 * 224 * 224;
    nchw_to_nhwc<<<dim3((unsigned)((total + 255) / 256)), 256, 0, stream>>>(
        (const float*)d_in[0], bx, 32, 3, 224, 224);
  }

  // ---- conv1 7x7 s2 + BN + ReLU (generic path, Cin=3) -> [112,112,64] ----
  {
    const int Cout = 64;
    const int Kpad = (3 * 7 * 7 + 31) & ~31;     // 160
    const int ntiles = Cout / 16;
    const size_t packElems = (size_t)(Kpad >> 5) * ntiles * 512;
    pack_w<3, 7, 7><<<dim3((unsigned)((packElems + 255) / 256)), 256, 0, stream>>>(
        (const float*)d_in[1], packb, Cout, ntiles);
    const int OH = 112, OW = 112, M = 32 * OH * OW;
    dim3 grid((unsigned)((M + 15) / 16), (unsigned)(Cout / 64));
    conv_bn_wmma_gen<3, 7, 7><<<grid, 32, 0, stream>>>(
        bx, packb,
        (const float*)d_in[2], (const float*)d_in[3],
        (const float*)d_in[4], (const float*)d_in[5],
        act0, 32, 224, 224, OH, OW, Cout, 2, 3, 1, ntiles);
  }
  // ---- maxpool 3x3 s2 -> [56,56,64] ----
  {
    size_t total = 32ull * 56 * 56 * 64;
    maxpool3s2_k<<<dim3((unsigned)((total + 255) / 256)), 256, 0, stream>>>(
        act0, act1, 32, 112, 112, 64, 56, 56);
  }

  // ---- stage 2 (64ch, 56x56) ----
  conv_layer<64,3>(d_in,  6, act1, 56, 56, 64, 1, 1, nullptr, 1, act2, packb, stream);
  conv_layer<64,3>(d_in, 11, act2, 56, 56, 64, 1, 1, act1,    1, act3, packb, stream);
  conv_layer<64,3>(d_in, 16, act3, 56, 56, 64, 1, 1, nullptr, 1, act2, packb, stream);
  conv_layer<64,3>(d_in, 21, act2, 56, 56, 64, 1, 1, act3,    1, act1, packb, stream);

  // ---- stage 3 (64->128, s2, 28x28) ----
  conv_layer<64,1> (d_in, 46, act1, 56, 56, 128, 2, 0, nullptr, 0, act3, packb, stream);
  conv_layer<64,3> (d_in, 26, act1, 56, 56, 128, 2, 1, nullptr, 1, act2, packb, stream);
  conv_layer<128,3>(d_in, 31, act2, 28, 28, 128, 1, 1, act3,    1, act0, packb, stream);
  conv_layer<128,3>(d_in, 36, act0, 28, 28, 128, 1, 1, nullptr, 1, act2, packb, stream);
  conv_layer<128,3>(d_in, 41, act2, 28, 28, 128, 1, 1, act0,    1, act1, packb, stream);

  // ---- stage 4 (128->256, s2, 14x14) ----
  conv_layer<128,1>(d_in, 71, act1, 28, 28, 256, 2, 0, nullptr, 0, act3, packb, stream);
  conv_layer<128,3>(d_in, 51, act1, 28, 28, 256, 2, 1, nullptr, 1, act2, packb, stream);
  conv_layer<256,3>(d_in, 56, act2, 14, 14, 256, 1, 1, act3,    1, act0, packb, stream);
  conv_layer<256,3>(d_in, 61, act0, 14, 14, 256, 1, 1, nullptr, 1, act2, packb, stream);
  conv_layer<256,3>(d_in, 66, act2, 14, 14, 256, 1, 1, act0,    1, act1, packb, stream);

  // ---- stage 5 (256->512, s2, 7x7) ----
  conv_layer<256,1>(d_in, 96, act1, 14, 14, 512, 2, 0, nullptr, 0, act3, packb, stream);
  conv_layer<256,3>(d_in, 76, act1, 14, 14, 512, 2, 1, nullptr, 1, act2, packb, stream);
  conv_layer<512,3>(d_in, 81, act2,  7,  7, 512, 1, 1, act3,    1, act0, packb, stream);
  conv_layer<512,3>(d_in, 86, act0,  7,  7, 512, 1, 1, nullptr, 1, act2, packb, stream);
  conv_layer<512,3>(d_in, 91, act2,  7,  7, 512, 1, 1, act0,    1, act1, packb, stream);

  // ---- avgpool 7x7 -> [32,512] ----
  avgpool7<<<dim3((unsigned)((32ull * 512 + 255) / 256)), 256, 0, stream>>>(
      act1, pooled, 32, 512);

  // ---- FC 512 -> 1000 (+bias), fp32 out ----
  {
    const int ntiles = 63;                       // N padded to 1008
    const size_t packElems = (size_t)(512 >> 5) * ntiles * 512;
    pack_w<512, 1, 1><<<dim3((unsigned)((packElems + 255) / 256)), 256, 0, stream>>>(
        (const float*)d_in[101], packb, 1000, ntiles);
    fc_wmma<<<dim3(2, (unsigned)63), 32, 0, stream>>>(
        pooled, packb, (const float*)d_in[102], (float*)d_out, 32, 512, 1000, 63);
  }
}
